// HGT_20813411516824
// MI455X (gfx1250) — matrix operand order repaired and checked by reading, not compiled, not run on hardware
//
#include <hip/hip_runtime.h>
#include <hip/hip_bf16.h>
#include <math.h>

// ---------------- problem constants (match reference) ----------------
#define N1c 6000
#define N2c 6000
#define HIDc 64
#define Hc 8
#define Lc 2
#define Ec 200000
#define ESc 150000

// ---------------- WMMA types ----------------
typedef __attribute__((ext_vector_type(16))) __bf16 v16bf;
typedef __attribute__((ext_vector_type(8)))  __bf16 v8bf;
typedef __attribute__((ext_vector_type(8)))  float  v8f;
typedef __attribute__((ext_vector_type(4)))  float  v4f;

// compile-time GEMM variants
#define GF_GELU_A 1   // apply exact gelu to A elements before matmul
#define GF_RELU   2   // relu(acc + bias)
#define GF_SIG    4   // alpha = sigmoid(alpha_ptr[idx])

// Per-lane WMMA operand layouts per CDNA5 ISA 7.12.2 (wave32):
//   A (16x32 bf16): lane<16 -> M=lane,   K = k0 + {0..7, 16..23}
//                   lane>=16-> M=lane-16,K = k0 + {8..15,24..31}
//   B (32x16 bf16): lane<16 -> N=lane,   K = k0 + {0..7, 16..23}
//                   lane>=16-> N=lane-16,K = k0 + {8..15,24..31}
//   C/D f32:        value r -> M = m0 + 8*(lane>=16) + r, N = n0 + (lane&15)

// =====================================================================
// GEMM flavor 1: A is f32 (loaded + optional gelu + cvt), B is bf16 [N,K].
// Used where A is consumed once (input proj) or needs fused f32 gelu (finish).
// =====================================================================
template<int FLAGS, int KSTEPS>
__global__ __launch_bounds__(128) void hgt_gemm_fa(
    const float* __restrict__ A, int lda,
    const __bf16* __restrict__ B, int ldb,
    const float* __restrict__ bias,
    const float* __restrict__ prev, int ldp,
    float* __restrict__ C, int ldc,
    int M, int N,
    const float* __restrict__ alpha_ptr, int alpha_idx) {
  int wave   = threadIdx.x >> 5;
  int lane   = threadIdx.x & 31;
  int nwaves = blockDim.x >> 5;
  int m0 = blockIdx.x * 16;
  int n0 = (blockIdx.y * nwaves + wave) * 16;
  if (m0 >= M || n0 >= N) return;     // wave-uniform: EXEC stays all-1 for WMMA

  int half = lane >> 4, l16 = lane & 15;
  const float*  arow = A + (size_t)(m0 + l16) * lda;
  const __bf16* brow = B + (size_t)(n0 + l16) * ldb;

  v8f acc = (v8f){0.f,0.f,0.f,0.f,0.f,0.f,0.f,0.f};
  #pragma unroll
  for (int ks = 0; ks < KSTEPS; ++ks) {
    const int k0 = ks * 32;
    float av[16];
    const float* ap0 = arow + k0 + half * 8;
    *(v4f*)(av + 0)  = *(const v4f*)(ap0);
    *(v4f*)(av + 4)  = *(const v4f*)(ap0 + 4);
    *(v4f*)(av + 8)  = *(const v4f*)(ap0 + 16);
    *(v4f*)(av + 12) = *(const v4f*)(ap0 + 20);
    if constexpr (FLAGS & GF_GELU_A) {
      #pragma unroll
      for (int j = 0; j < 16; ++j) {
        float x = av[j];
        av[j] = 0.5f * x * (1.0f + erff(x * 0.70710678118654752f));
      }
    }
    v16bf aop;
    #pragma unroll
    for (int j = 0; j < 16; ++j) aop[j] = (__bf16)av[j];   // v_cvt_pk_bf16_f32

    const __bf16* bp = brow + k0 + half * 8;
    v8bf blo = *(const v8bf*)(bp);
    v8bf bhi = *(const v8bf*)(bp + 16);
    v16bf bop = __builtin_shufflevector(blo, bhi, 0,1,2,3,4,5,6,7,8,9,10,11,12,13,14,15);
    acc = __builtin_amdgcn_wmma_f32_16x16x32_bf16(
        false, aop, false, bop, (short)0, acc, false, false);
  }

  float alpha = 1.0f;
  if (alpha_ptr) {
    alpha = alpha_ptr[alpha_idx];
    if constexpr (FLAGS & GF_SIG) alpha = 1.0f / (1.0f + expf(-alpha));
  }
  int ccol = n0 + l16;
  float bval = bias ? bias[ccol] : 0.0f;
  #pragma unroll
  for (int r = 0; r < 8; ++r) {
    int crow = m0 + half * 8 + r;
    float t = acc[r] + bval;
    if constexpr (FLAGS & GF_RELU) t = fmaxf(t, 0.0f);
    float out = alpha * t;
    if (prev) out += (1.0f - alpha) * prev[(size_t)crow * ldp + ccol];
    C[(size_t)crow * ldc + ccol] = out;
  }
}

// =====================================================================
// GEMM flavor 2: A and B both bf16 (pre-packed) -> pure load+WMMA loop.
// NT tiles per wave share one A operand (used for the 6000x6000 GEMM).
// =====================================================================
template<int KSTEPS, int NT>
__global__ __launch_bounds__(128) void hgt_gemm_ba(
    const __bf16* __restrict__ A, int lda,
    const __bf16* __restrict__ B, int ldb,
    const float* __restrict__ bias,
    float* __restrict__ C, int ldc,
    int M, int N) {
  int wave   = threadIdx.x >> 5;
  int lane   = threadIdx.x & 31;
  int nwaves = blockDim.x >> 5;
  int m0 = blockIdx.x * 16;
  int n0 = (blockIdx.y * nwaves + wave) * (NT * 16);
  if (m0 >= M || n0 >= N) return;

  int half = lane >> 4, l16 = lane & 15;
  int  n1  = (NT == 2) ? ((n0 + 16 < N) ? (n0 + 16) : n0) : n0;   // clamp, store guarded
  bool st2 = (NT == 2) && (n0 + 16 < N);

  const __bf16* arow  = A + (size_t)(m0 + l16) * lda;
  const __bf16* brow0 = B + (size_t)(n0 + l16) * ldb;
  const __bf16* brow1 = B + (size_t)(n1 + l16) * ldb;

  v8f acc0 = (v8f){0.f,0.f,0.f,0.f,0.f,0.f,0.f,0.f};
  v8f acc1 = acc0;
  #pragma unroll
  for (int ks = 0; ks < KSTEPS; ++ks) {
    const int k0 = ks * 32;
    const __bf16* ap = arow + k0 + half * 8;
    v8bf alo = *(const v8bf*)(ap);
    v8bf ahi = *(const v8bf*)(ap + 16);
    v16bf aop = __builtin_shufflevector(alo, ahi, 0,1,2,3,4,5,6,7,8,9,10,11,12,13,14,15);

    const __bf16* bp0 = brow0 + k0 + half * 8;
    v8bf blo = *(const v8bf*)(bp0);
    v8bf bhi = *(const v8bf*)(bp0 + 16);
    v16bf bop = __builtin_shufflevector(blo, bhi, 0,1,2,3,4,5,6,7,8,9,10,11,12,13,14,15);
    acc0 = __builtin_amdgcn_wmma_f32_16x16x32_bf16(
        false, aop, false, bop, (short)0, acc0, false, false);

    if constexpr (NT == 2) {
      const __bf16* bp1 = brow1 + k0 + half * 8;
      v8bf clo = *(const v8bf*)(bp1);
      v8bf chi = *(const v8bf*)(bp1 + 16);
      v16bf cop = __builtin_shufflevector(clo, chi, 0,1,2,3,4,5,6,7,8,9,10,11,12,13,14,15);
      acc1 = __builtin_amdgcn_wmma_f32_16x16x32_bf16(
          false, aop, false, cop, (short)0, acc1, false, false);
    }
  }

  auto epi = [&](const v8f& acc, int nbase) {
    int ccol = nbase + l16;
    float bval = bias ? bias[ccol] : 0.0f;
    #pragma unroll
    for (int r = 0; r < 8; ++r) {
      int crow = m0 + half * 8 + r;
      C[(size_t)crow * ldc + ccol] = acc[r] + bval;
    }
  };
  epi(acc0, n0);
  if (st2) epi(acc1, n1);
}

// ===================== small helper kernels =====================
__global__ void hgt_fill_kernel(float* p, float v, int n) {
  int i = blockIdx.x * blockDim.x + threadIdx.x;
  if (i < n) p[i] = v;
}

// f32 -> bf16 row-major pack (activation / Em / Ed reuse amortization)
__global__ void hgt_pack_bf16(const float* __restrict__ in, __bf16* __restrict__ out, int n) {
  int i = blockIdx.x * blockDim.x + threadIdx.x;
  if (i < n) out[i] = (__bf16)in[i];
}

// outT[n*K + k] = (bf16) in[k*N + n]   (weight pre-transpose + pack, tiny)
__global__ void hgt_transpose_bf(const float* __restrict__ in, __bf16* __restrict__ out,
                                 int K, int N) {
  int i = blockIdx.x * blockDim.x + threadIdx.x;
  if (i >= K * N) return;
  int k = i / N, n = i - k * N;
  out[(size_t)n * K + k] = (__bf16)in[i];
}

// Fold block-diagonal relation R[h][d][e] into projection weight/bias; write
// the weight TRANSPOSED + bf16-packed ([he][k]) for the contiguous-B path:
//   WoutT[he][k] = sum_d W[k][h*8+d] * R[h][d][e]
__global__ void hgt_fold_kernel(const float* __restrict__ W, const float* __restrict__ bin,
                                const float* __restrict__ R,
                                __bf16* __restrict__ WoutT, float* __restrict__ bout) {
  int he = threadIdx.x;            // 0..63
  int h = he >> 3, e = he & 7;
  int k = blockIdx.x;              // 0..64 (64 == bias row)
  float s = 0.0f;
  if (k < 64) {
    #pragma unroll
    for (int d = 0; d < 8; ++d) s += W[k * 64 + h * 8 + d] * R[(h * 8 + d) * 8 + e];
    WoutT[(size_t)he * 64 + k] = (__bf16)s;
  } else {
    #pragma unroll
    for (int d = 0; d < 8; ++d) s += bin[h * 8 + d] * R[(h * 8 + d) * 8 + e];
    bout[he] = s;
  }
}

__device__ __forceinline__ void hgt_atomicMaxF(float* addr, float val) {
  // sign-trick float max; correct for +-0 and init -inf
  if (__float_as_int(val) >= 0) atomicMax((int*)addr, __float_as_int(val));
  else                          atomicMin((unsigned int*)addr, __float_as_uint(val));
}

// alpha[e][h] = <q[d],kk[s]>_h * pr[h]/sqrt(8);  amax[d][h] = segment max
__global__ void hgt_edge_alpha(const float* __restrict__ kk, const float* __restrict__ q,
                               const int* __restrict__ ei, const float* __restrict__ pr,
                               float* __restrict__ alpha, float* __restrict__ amax, int E) {
  int e = blockIdx.x * blockDim.x + threadIdx.x;
  if (e >= E) return;
  int si = ei[e], di = ei[E + e];
  const float* qd = q  + (size_t)di * 64;
  const float* ks = kk + (size_t)si * 64;
  const float inv = 0.35355339059327373f;   // 1/sqrt(DH)
  #pragma unroll
  for (int h = 0; h < 8; ++h) {
    float s = 0.0f;
    #pragma unroll
    for (int j = 0; j < 8; ++j) s += qd[h * 8 + j] * ks[h * 8 + j];
    s *= pr[h] * inv;
    alpha[(size_t)e * 8 + h] = s;
    hgt_atomicMaxF(&amax[di * 8 + h], s);
  }
}

// ex = exp(alpha - amax[d]);  den[d][h] = segment sum
__global__ void hgt_edge_exp(const int* __restrict__ ei, float* __restrict__ alpha,
                             const float* __restrict__ amax, float* __restrict__ den, int E) {
  int e = blockIdx.x * blockDim.x + threadIdx.x;
  if (e >= E) return;
  int di = ei[E + e];
  #pragma unroll
  for (int h = 0; h < 8; ++h) {
    float ex = expf(alpha[(size_t)e * 8 + h] - amax[di * 8 + h]);
    alpha[(size_t)e * 8 + h] = ex;
    atomicAdd(&den[di * 8 + h], ex);
  }
}

// o[d] += vv[s] * ex/(den[d]+eps)
__global__ void hgt_edge_scatter(const int* __restrict__ ei, const float* __restrict__ vv,
                                 const float* __restrict__ ex, const float* __restrict__ den,
                                 float* __restrict__ o, int E) {
  int e = blockIdx.x * blockDim.x + threadIdx.x;
  if (e >= E) return;
  int si = ei[e], di = ei[E + e];
  const float* vs = vv + (size_t)si * 64;
  #pragma unroll
  for (int h = 0; h < 8; ++h) {
    float w = ex[(size_t)e * 8 + h] / (den[di * 8 + h] + 1e-16f);
    #pragma unroll
    for (int j = 0; j < 8; ++j)
      atomicAdd(&o[(size_t)di * 64 + h * 8 + j], vs[h * 8 + j] * w);
  }
}

// dst[r, coloff + c] += coef[ci] * src[r, c]   (src is [rows,64])
__global__ void hgt_accum(float* __restrict__ dst, int ld, int coloff,
                          const float* __restrict__ src, const float* __restrict__ coef,
                          int ci, int rows) {
  int i = blockIdx.x * blockDim.x + threadIdx.x;
  if (i >= rows * 64) return;
  int r = i >> 6, c = i & 63;
  dst[(size_t)r * ld + coloff + c] += coef[ci] * src[i];
}

__global__ void hgt_gather(const int* __restrict__ se, const float* __restrict__ yall,
                           float* __restrict__ y, int ES, int n2) {
  int i = blockIdx.x * blockDim.x + threadIdx.x;
  if (i < ES) y[i] = yall[(size_t)se[i] * n2 + se[ES + i]];
}

// =====================================================================
extern "C" void kernel_launch(void* const* d_in, const int* in_sizes, int n_in,
                              void* d_out, int out_size, void* d_ws, size_t ws_size,
                              hipStream_t stream) {
  (void)in_sizes; (void)n_in; (void)out_size; (void)ws_size;

  const float* xe1  = (const float*)d_in[2];
  const float* xe2  = (const float*)d_in[3];
  const int*   ei12 = (const int*)d_in[4];
  const int*   ei21 = (const int*)d_in[5];
  const int*   se   = (const int*)d_in[6];
  const float* lxw  = (const float*)d_in[9];
  const float* lxb  = (const float*)d_in[10];
  const float* xb1  = (const float*)d_in[11];
  const float* xb2  = (const float*)d_in[12];
  const float* lbw  = (const float*)d_in[13];
  const float* lbb  = (const float*)d_in[14];
  const float* att  = (const float*)d_in[15];
  const float* att2 = (const float*)d_in[16];
  const float* CW[2] = {(const float*)d_in[17], (const float*)d_in[25]};
  const float* CB[2] = {(const float*)d_in[18], (const float*)d_in[26]};
  const float* AW[2] = {(const float*)d_in[19], (const float*)d_in[27]};
  const float* AB[2] = {(const float*)d_in[20], (const float*)d_in[28]};
  const float* SK[2] = {(const float*)d_in[21], (const float*)d_in[29]};
  const float* AR[2] = {(const float*)d_in[22], (const float*)d_in[30]};
  const float* MR[2] = {(const float*)d_in[23], (const float*)d_in[31]};
  const float* PR[2] = {(const float*)d_in[24], (const float*)d_in[32]};

  // -------- workspace carve-up (f32 section then bf16 section, ~37 MB) ----
  float* ws = (float*)d_ws;
  const size_t SH = (size_t)N1c * HIDc;                 // 384000
  float* h1 = ws;        float* h2 = h1 + SH;
  float* g1 = h2 + SH;   float* g2 = g1 + SH;
  float* q1 = g2 + SH;   float* q2 = q1 + SH;
  float* kk1= q2 + SH;   float* kk2= kk1 + SH;
  float* vv1= kk2 + SH;  float* vv2= vv1 + SH;
  float* o1 = vv2 + SH;  float* o2 = o1 + SH;
  float* alB  = o2 + SH;                                // E*8
  float* amax = alB + (size_t)Ec * Hc;                  // 48000
  float* den  = amax + (size_t)N1c * Hc;
  float* Em   = den  + (size_t)N1c * Hc;                // 6000*128
  float* Ed   = Em   + (size_t)N1c * 128;
  float* fB   = Ed   + (size_t)N2c * 128;               // 16 x 64 folded biases (f32)
  __bf16* bws  = (__bf16*)(fB + (size_t)16 * 64);       // ---- bf16 section ----
  __bf16* fWT  = bws;                                   // 16 x 64x64 folded k/v (T)
  __bf16* lxwT = fWT  + (size_t)16 * 4096;              // 2 x 64x128
  __bf16* lbwT = lxwT + (size_t)2 * 8192;               // 2 x 64x128
  __bf16* qwT  = lbwT + (size_t)2 * 8192;               // 8 x 64x64
  __bf16* awT  = qwT  + (size_t)8 * 4096;               // 8 x 64x64
  __bf16* hB1  = awT  + (size_t)8 * 4096;               // 6000x64 packed activations
  __bf16* hB2  = hB1  + SH;
  __bf16* EmB  = hB2  + SH;                             // 6000x128 packed
  __bf16* EdB  = EmB  + (size_t)N1c * 128;

  auto F = [&](float* p, float v, size_t n) {
    hgt_fill_kernel<<<dim3((unsigned)((n + 255) / 256)), dim3(256), 0, stream>>>(p, v, (int)n);
  };
  auto P = [&](const float* in, __bf16* out, size_t n) {
    hgt_pack_bf16<<<dim3((unsigned)((n + 255) / 256)), dim3(256), 0, stream>>>(in, out, (int)n);
  };
  auto T = [&](const float* in, __bf16* out, int K, int N) {
    hgt_transpose_bf<<<dim3((K * N + 255) / 256), dim3(256), 0, stream>>>(in, out, K, N);
  };
  auto EDGE = [&](const float* kk, const float* vv, const float* q, const int* ei,
                  const float* pr, float* o) {
    F(amax, -INFINITY, (size_t)N1c * Hc);
    F(den, 0.0f, (size_t)N1c * Hc);
    F(o, 0.0f, SH);
    int gb = (Ec + 255) / 256;
    hgt_edge_alpha  <<<gb, 256, 0, stream>>>(kk, q, ei, pr, alB, amax, Ec);
    hgt_edge_exp    <<<gb, 256, 0, stream>>>(ei, alB, amax, den, Ec);
    hgt_edge_scatter<<<gb, 256, 0, stream>>>(ei, vv, alB, den, o, Ec);
  };

  // -------- pre-transpose+pack all weights; fold a_rel/m_rel into k/v ----
  T(lxw,        lxwT,        128, 64);
  T(lxw + 8192, lxwT + 8192, 128, 64);
  T(lbw,        lbwT,        128, 64);
  T(lbw + 8192, lbwT + 8192, 128, 64);
  for (int br = 0; br < 2; ++br)
    for (int l = 0; l < Lc; ++l)
      for (int nt = 0; nt < 2; ++nt) {
        int idx = (br * 2 + l) * 2 + nt;
        T(CW[br] + ((l * 3 + 1) * 2 + nt) * 4096, qwT + (size_t)idx * 4096, 64, 64);
        T(AW[br] + (l * 2 + nt) * 4096,           awT + (size_t)idx * 4096, 64, 64);
        int fk = idx * 2, fv = fk + 1;
        hgt_fold_kernel<<<65, 64, 0, stream>>>(
            CW[br] + ((l * 3 + 0) * 2 + nt) * 4096, CB[br] + ((l * 3 + 0) * 2 + nt) * 64,
            AR[br] + (l * 2 + nt) * 512, fWT + (size_t)fk * 4096, fB + (size_t)fk * 64);
        hgt_fold_kernel<<<65, 64, 0, stream>>>(
            CW[br] + ((l * 3 + 2) * 2 + nt) * 4096, CB[br] + ((l * 3 + 2) * 2 + nt) * 64,
            MR[br] + (l * 2 + nt) * 512, fWT + (size_t)fv * 4096, fB + (size_t)fv * 64);
      }

  // -------- input projections (0*lin(x) term is exactly zero -> skipped) ----
  dim3 g375(375, 1), blk(128);
  hgt_gemm_fa<GF_RELU,4><<<g375, blk, 0, stream>>>(xe1, 128, lxwT,        128, lxb,      nullptr, 0, h1, 64, N1c, 64, att, 1);
  hgt_gemm_fa<GF_RELU,4><<<g375, blk, 0, stream>>>(xe2, 128, lxwT + 8192, 128, lxb + 64, nullptr, 0, h2, 64, N2c, 64, att, 1);
  hgt_gemm_fa<GF_RELU,4><<<g375, blk, 0, stream>>>(xb1, 128, lbwT,        128, lbb,      nullptr, 0, g1, 64, N1c, 64, nullptr, 0);
  hgt_gemm_fa<GF_RELU,4><<<g375, blk, 0, stream>>>(xb2, 128, lbwT + 8192, 128, lbb + 64, nullptr, 0, g2, 64, N2c, 64, nullptr, 0);

  F(Em, 0.0f, (size_t)N1c * 128);
  F(Ed, 0.0f, (size_t)N2c * 128);

  // -------- two branches x two HGT conv layers --------
  for (int br = 0; br < 2; ++br) {
    float* hh1 = br ? g1 : h1;
    float* hh2 = br ? g2 : h2;
    for (int l = 0; l < Lc; ++l) {
      const float* cb = CB[br];
      int i0 = (br * 2 + l) * 2 + 0, i1 = i0 + 1;   // per (br,l,nt) weight index
      // pack current activations once; reused by q/kk/vv (3 GEMMs each)
      P(hh1, hB1, SH);
      P(hh2, hB2, SH);
      // q projections
      hgt_gemm_ba<2,1><<<g375, blk, 0, stream>>>(hB1, 64, qwT + (size_t)i0 * 4096, 64,
          cb + ((l * 3 + 1) * 2 + 0) * 64, q1, 64, N1c, 64);
      hgt_gemm_ba<2,1><<<g375, blk, 0, stream>>>(hB2, 64, qwT + (size_t)i1 * 4096, 64,
          cb + ((l * 3 + 1) * 2 + 1) * 64, q2, 64, N2c, 64);
      // folded k' = k@ar, v' = v@mr projections
      hgt_gemm_ba<2,1><<<g375, blk, 0, stream>>>(hB1, 64, fWT + (size_t)(i0*2+0) * 4096, 64,
          fB + (size_t)(i0*2+0) * 64, kk1, 64, N1c, 64);
      hgt_gemm_ba<2,1><<<g375, blk, 0, stream>>>(hB1, 64, fWT + (size_t)(i0*2+1) * 4096, 64,
          fB + (size_t)(i0*2+1) * 64, vv1, 64, N1c, 64);
      hgt_gemm_ba<2,1><<<g375, blk, 0, stream>>>(hB2, 64, fWT + (size_t)(i1*2+0) * 4096, 64,
          fB + (size_t)(i1*2+0) * 64, kk2, 64, N2c, 64);
      hgt_gemm_ba<2,1><<<g375, blk, 0, stream>>>(hB2, 64, fWT + (size_t)(i1*2+1) * 4096, 64,
          fB + (size_t)(i1*2+1) * 64, vv2, 64, N2c, 64);
      // gather-attend-scatter, both directions
      EDGE(kk1, vv1, q2, ei12, PR[br] + (l * 2 + 0) * 8, o2);   // n1 -> n2
      EDGE(kk2, vv2, q1, ei21, PR[br] + (l * 2 + 1) * 8, o1);   // n2 -> n1
      // finish: h = sig(skip)*(gelu(o)@aw + ab) + (1-sig(skip))*h   (in-place safe)
      hgt_gemm_fa<GF_GELU_A|GF_SIG,2><<<g375, blk, 0, stream>>>(o1, 64,
          awT + (size_t)i0 * 4096, 64, AB[br] + (l * 2 + 0) * 64,
          hh1, 64, hh1, 64, N1c, 64, SK[br], l * 2 + 0);
      hgt_gemm_fa<GF_GELU_A|GF_SIG,2><<<g375, blk, 0, stream>>>(o2, 64,
          awT + (size_t)i1 * 4096, 64, AB[br] + (l * 2 + 1) * 64,
          hh2, 64, hh2, 64, N2c, 64, SK[br], l * 2 + 1);
      // Em/Ed += att2[br] * h  at columns [l*64, l*64+64)
      hgt_accum<<<1500, 256, 0, stream>>>(Em, 128, l * 64, hh1, att2, br, N1c);
      hgt_accum<<<1500, 256, 0, stream>>>(Ed, 128, l * 64, hh2, att2, br, N2c);
    }
  }

  // -------- y_all = Em @ Ed^T: pack once to bf16, then pure-WMMA GEMM ----
  P(Em, EmB, (size_t)N1c * 128);
  P(Ed, EdB, (size_t)N2c * 128);
  float* yall = (float*)d_out + ESc;
  hgt_gemm_ba<4,2><<<dim3(375, 47), blk, 0, stream>>>(EmB, 128, EdB, 128, nullptr,
      yall, N2c, N1c, N2c);
  hgt_gather<<<(ESc + 255) / 256, 256, 0, stream>>>(se, yall, (float*)d_out, ESc, N2c);
}
